// MambaEncoder_4801773437302
// MI455X (gfx1250) — matrix operand order repaired and checked by reading, not compiled, not run on hardware
//
#include <hip/hip_runtime.h>

// ---------------------------------------------------------------------------
// MambaEncoder for MI455X (gfx1250, wave32).
// GEMMs: v_wmma_f32_16x16x32_bf16 (fp32 accumulate), tiles staged into LDS
// with GLOBAL_LOAD_ASYNC_TO_LDS_B128 (ASYNCcnt + s_wait_asynccnt) when the
// toolchain exposes the builtins, then converted fp32->bf16 in LDS.
// Selective scan: one thread per (batch, channel), 16 fp32 states in regs.
// Workspace usage: ~272 MB.
// ---------------------------------------------------------------------------

typedef __attribute__((ext_vector_type(16))) __bf16 v16bf;
typedef __attribute__((ext_vector_type(8)))  float  v8f;
typedef __attribute__((ext_vector_type(4)))  float  v4f;
typedef __attribute__((ext_vector_type(4)))  int    v4i;

static constexpr int Bb  = 4;
static constexpr int Ll  = 2048;
static constexpr int INq = 512;
static constexpr int Dd  = 1024;
static constexpr int NL  = 2;
static constexpr int Ss  = 16;
static constexpr int Kc  = 8;
static constexpr int DI  = 2048;           // d_inner
static constexpr int DTR = 64;             // dt_rank
static constexpr int XD  = DTR + 2 * Ss;   // 96
static constexpr int ROWS = Bb * Ll;       // 8192

#if defined(__gfx1250__)
#  if __has_builtin(__builtin_amdgcn_global_load_async_to_lds_b128) && \
      __has_builtin(__builtin_amdgcn_s_wait_asynccnt)
#    define USE_ASYNC_LDS 1
#  else
#    define USE_ASYNC_LDS 0
#    warning "gfx1250: async-to-LDS builtins unavailable; using sync staging fallback"
#  endif
#else
#  define USE_ASYNC_LDS 0
#endif

#if USE_ASYNC_LDS
// Builtin signature (from compiler diagnostic): param0 is a non-const
// addrspace(1) v4i* (global source); param1 is the LDS destination.
typedef __attribute__((address_space(1))) v4i* gv4i_p;
typedef __attribute__((address_space(3))) v4i* lv4i_p;
#endif

__device__ __forceinline__ __bf16 f2bf(float f) {
  // round-to-nearest-even fp32 -> bf16, pure bit math
  unsigned u = __builtin_bit_cast(unsigned, f);
  u += 0x7fffu + ((u >> 16) & 1u);
  unsigned short hs = (unsigned short)(u >> 16);
  return __builtin_bit_cast(__bf16, hs);
}

__device__ __forceinline__ float sigmoidf_(float x) {
  return 1.0f / (1.0f + __expf(-x));
}

// ---------------------------------------------------------------------------
// GEMM: C[m,n] = sum_k A[m,k] * W[n,k]  (+ epilogue)
//   EPI 0: none | 1: +bias[n] | 2: softplus(.+bias[n]) | 3: +resid[m,n]
// Block: 128 threads = 4 waves. Block tile 64x64, wave tile 16x64.
// Requirements met by all call sites: M%64==0, K%32==0, lda/ldw rows 16B
// aligned. Only N may be ragged (x_proj N=96) -> zero-fill guard on W tile.
// ---------------------------------------------------------------------------
template <int EPI>
__global__ __launch_bounds__(128)
void gemm_bf16(const float* __restrict__ A, int lda,
               const float* __restrict__ W, int ldw,
               const float* __restrict__ bias,
               const float* __restrict__ resid,
               float* __restrict__ C, int ldc,
               int M, int N, int K)
{
  __shared__ float  fA[64 * 36];   // raw fp32 tile (async DMA target), 144B rows
  __shared__ float  fW[64 * 36];
  __shared__ __bf16 sA[64 * 36];   // bf16 tiles in WMMA gather layout
  __shared__ __bf16 sW[64 * 36];

  const int tid  = threadIdx.x;
  const int wv   = tid >> 5;       // wave id 0..3 -> M sub-tile
  const int lane = tid & 31;
  const int r    = lane & 15;
  const int half = lane >> 4;
  const int m0   = blockIdx.y * 64;
  const int n0   = blockIdx.x * 64;

  v8f acc[4] = {};

  for (int k0 = 0; k0 < K; k0 += 32) {
    // ---- stage raw fp32 tiles --------------------------------------------
#if USE_ASYNC_LDS
    for (int i = tid; i < 64 * 8; i += 128) {       // 512 b128 xfers / tile
      const int row = i >> 3;
      const int c4  = (i & 7) << 2;                 // 4-float column group
      __builtin_amdgcn_global_load_async_to_lds_b128(
          (gv4i_p)(A + (size_t)(m0 + row) * lda + k0 + c4),
          (lv4i_p)(&fA[row * 36 + c4]), 0, 0);
      if (n0 + row < N) {
        __builtin_amdgcn_global_load_async_to_lds_b128(
            (gv4i_p)(W + (size_t)(n0 + row) * ldw + k0 + c4),
            (lv4i_p)(&fW[row * 36 + c4]), 0, 0);
      } else {
        *(v4f*)&fW[row * 36 + c4] = (v4f)(0.0f);
      }
    }
    __builtin_amdgcn_s_wait_asynccnt(0);            // my DMAs landed in LDS
    __syncthreads();                                // everyone's DMAs visible
#else
    for (int i = tid; i < 64 * 8; i += 128) {
      const int row = i >> 3;
      const int c4  = (i & 7) << 2;
      *(v4f*)&fA[row * 36 + c4] =
          *(const v4f*)(A + (size_t)(m0 + row) * lda + k0 + c4);
      v4f wq = (v4f)(0.0f);
      if (n0 + row < N)
        wq = *(const v4f*)(W + (size_t)(n0 + row) * ldw + k0 + c4);
      *(v4f*)&fW[row * 36 + c4] = wq;
    }
    if (k0 + 32 < K)
      __builtin_prefetch(&A[(size_t)(m0 + (tid >> 1)) * lda + k0 + 32], 0, 1);
    __syncthreads();
#endif

    // ---- convert fp32 -> bf16 in LDS -------------------------------------
    for (int i = tid; i < 64 * 32; i += 128) {
      const int row = i >> 5, col = i & 31;
      sA[row * 36 + col] = f2bf(fA[row * 36 + col]);
      sW[row * 36 + col] = f2bf(fW[row * 36 + col]);
    }
    __syncthreads();

    // ---- WMMA -------------------------------------------------------------
    // A fragment: 16x32 bf16. lanes 0-15: rows M=0..15, K 0..15;
    // lanes 16-31: same rows, K 16..31.
    v16bf af;
    const int arow = wv * 16 + r;
    #pragma unroll
    for (int v = 0; v < 8; ++v) {
      const int kk = 2 * v + 16 * half;
      af[2 * v]     = sA[arow * 36 + kk];
      af[2 * v + 1] = sA[arow * 36 + kk + 1];
    }

    #pragma unroll
    for (int j = 0; j < 4; ++j) {
      // B fragment: 32x16 bf16. lane = N%16; lanes 0-15 hold K 0..15,
      // lanes 16-31 hold K 16..31, two K per VGPR.
      v16bf bf;
      const int bcol = j * 16 + r;
      #pragma unroll
      for (int v = 0; v < 8; ++v) {
        const int kk = 2 * v + 16 * half;
        bf[2 * v]     = sW[bcol * 36 + kk];
        bf[2 * v + 1] = sW[bcol * 36 + kk + 1];
      }
      acc[j] = __builtin_amdgcn_wmma_f32_16x16x32_bf16(
          false, af, false, bf, (short)0, acc[j], false, false);
    }
  }

  // C layout: VGPR v, lanes 0-15 -> M=v, N=lane; lanes 16-31 -> M=8+v.
  #pragma unroll
  for (int j = 0; j < 4; ++j) {
    const int n = n0 + j * 16 + r;
    if (n >= N) continue;
    #pragma unroll
    for (int v = 0; v < 8; ++v) {
      const int m = m0 + wv * 16 + half * 8 + v;
      float val = acc[j][v];
      if (EPI == 1 || EPI == 2) val += bias[n];
      if (EPI == 2) val = (val > 20.0f) ? val : __logf(1.0f + __expf(val));
      if (EPI == 3) val += resid[(size_t)m * ldc + n];
      C[(size_t)m * ldc + n] = val;
    }
  }
}

// ---------------------------------------------------------------------------
// LayerNorm: one block per row of Dn elements.
// ---------------------------------------------------------------------------
__global__ __launch_bounds__(256)
void layernorm_k(const float* __restrict__ in, const float* __restrict__ w,
                 const float* __restrict__ b, float* __restrict__ out, int Dn)
{
  __shared__ float red[256];
  const int row = blockIdx.x;
  const int tid = threadIdx.x;
  const float* xr = in + (size_t)row * Dn;

  float s = 0.f;
  for (int i = tid; i < Dn; i += 256) s += xr[i];
  red[tid] = s; __syncthreads();
  for (int off = 128; off > 0; off >>= 1) {
    if (tid < off) red[tid] += red[tid + off];
    __syncthreads();
  }
  const float mean = red[0] / (float)Dn;
  __syncthreads();

  float v = 0.f;
  for (int i = tid; i < Dn; i += 256) { float d = xr[i] - mean; v += d * d; }
  red[tid] = v; __syncthreads();
  for (int off = 128; off > 0; off >>= 1) {
    if (tid < off) red[tid] += red[tid + off];
    __syncthreads();
  }
  const float rstd = rsqrtf(red[0] / (float)Dn + 1e-5f);

  float* yr = out + (size_t)row * Dn;
  for (int i = tid; i < Dn; i += 256)
    yr[i] = (xr[i] - mean) * rstd * w[i] + b[i];
}

// ---------------------------------------------------------------------------
// Causal depthwise conv (K=8) + SiLU. Reads xh half of xz, writes xh buffer.
// ---------------------------------------------------------------------------
__global__ __launch_bounds__(256)
void conv_silu_k(const float* __restrict__ xz, const float* __restrict__ cw,
                 const float* __restrict__ cb, float* __restrict__ xh)
{
  const size_t idx = (size_t)blockIdx.x * 256 + threadIdx.x;
  const int c = (int)(idx % DI);
  const int l = (int)((idx / DI) % Ll);
  const int b = (int)(idx / ((size_t)DI * Ll));
  float s = cb[c];
  #pragma unroll
  for (int k = 0; k < Kc; ++k) {
    const int ll = l - (Kc - 1) + k;
    if (ll >= 0)
      s += cw[c * Kc + k] * xz[(size_t)(b * Ll + ll) * (2 * DI) + c];
  }
  xh[idx] = s * sigmoidf_(s);   // SiLU
}

// ---------------------------------------------------------------------------
// Selective scan fused with +u*D and *silu(z) gate. One thread per channel,
// 16 states in registers. delta lives in the xh-half of xz (written there by
// dt_proj with ldc=2*DI), z in the upper half. y overwrites xh in place.
// ---------------------------------------------------------------------------
__global__ __launch_bounds__(256)
void scan_k(const float* __restrict__ xz,     // [ROWS, 2*DI]: delta | z
            float* __restrict__ xh,           // in: silu(conv); out: gated y
            const float* __restrict__ xdbl,   // [ROWS, 96]: dt | B | C
            const float* __restrict__ A_log,  // [DI, S]
            const float* __restrict__ Dp)     // [DI]
{
  const int tid = threadIdx.x;
  const int b   = blockIdx.x / (DI / 256);
  const int c   = (blockIdx.x % (DI / 256)) * 256 + tid;

  float Ar[Ss];
  #pragma unroll
  for (int s = 0; s < Ss; ++s) Ar[s] = -__expf(A_log[c * Ss + s]);
  float hst[Ss] = {};
  const float dp = Dp[c];

  __shared__ float sB[Ss], sC[Ss];

  for (int l = 0; l < Ll; ++l) {
    const size_t rowL = (size_t)b * Ll + l;
    if (tid < Ss)            sB[tid]      = xdbl[rowL * XD + DTR + tid];
    else if (tid < 2 * Ss)   sC[tid - Ss] = xdbl[rowL * XD + DTR + Ss + (tid - Ss)];
    __syncthreads();

    const float d  = xz[rowL * (2 * DI) + c];         // softplus'd delta
    const float u  = xh[rowL * DI + c];
    const float du = d * u;
    float y = 0.f;
    #pragma unroll
    for (int s = 0; s < Ss; ++s) {
      hst[s] = __expf(d * Ar[s]) * hst[s] + du * sB[s];
      y += hst[s] * sC[s];
    }
    y += u * dp;
    const float z = xz[rowL * (2 * DI) + DI + c];
    y *= z * sigmoidf_(z);                            // gate
    xh[rowL * DI + c] = y;
    __syncthreads();
  }
}

// ---------------------------------------------------------------------------
extern "C" void kernel_launch(void* const* d_in, const int* in_sizes, int n_in,
                              void* d_out, int out_size, void* d_ws, size_t ws_size,
                              hipStream_t stream)
{
  const float* x            = (const float*)d_in[0];
  const float* input_proj_w = (const float*)d_in[1];
  const float* input_proj_b = (const float*)d_in[2];
  const float* norm1_w      = (const float*)d_in[3];
  const float* norm1_b      = (const float*)d_in[4];
  const float* in_proj_w    = (const float*)d_in[5];
  const float* conv_w       = (const float*)d_in[6];
  const float* conv_b       = (const float*)d_in[7];
  const float* x_proj_w     = (const float*)d_in[8];
  const float* dt_proj_w    = (const float*)d_in[9];
  const float* dt_proj_b    = (const float*)d_in[10];
  const float* A_log        = (const float*)d_in[11];
  const float* D_param      = (const float*)d_in[12];
  const float* out_proj_w   = (const float*)d_in[13];
  const float* norm_f_w     = (const float*)d_in[14];
  const float* norm_f_b     = (const float*)d_in[15];

  (void)in_sizes; (void)n_in; (void)out_size; (void)ws_size;

  // workspace carve-up (~272 MB)
  char* base = (char*)d_ws;
  auto take = [&](size_t bytes) -> float* {
    char* p = base;
    base += (bytes + 255) & ~(size_t)255;
    return (float*)p;
  };
  float* h    = take((size_t)ROWS * Dd * 4);        // residual stream
  float* hn   = take((size_t)ROWS * Dd * 4);        // layernorm output
  float* xzb  = take((size_t)ROWS * 2 * DI * 4);    // xz; later delta|z
  float* xhb  = take((size_t)ROWS * DI * 4);        // conv/silu out; later y
  float* xdbl = take((size_t)ROWS * XD * 4);        // dt | B | C

  const dim3 blk(128);

  // h = x @ input_proj_w^T + input_proj_b
  gemm_bf16<1><<<dim3(Dd / 64, ROWS / 64), blk, 0, stream>>>(
      x, INq, input_proj_w, INq, input_proj_b, nullptr, h, Dd,
      ROWS, Dd, INq);

  for (int i = 0; i < NL; ++i) {
    layernorm_k<<<ROWS, 256, 0, stream>>>(
        h, norm1_w + i * Dd, norm1_b + i * Dd, hn, Dd);

    // xz = hn @ in_proj_w^T
    gemm_bf16<0><<<dim3((2 * DI) / 64, ROWS / 64), blk, 0, stream>>>(
        hn, Dd, in_proj_w + (size_t)i * 2 * DI * Dd, Dd, nullptr, nullptr,
        xzb, 2 * DI, ROWS, 2 * DI, Dd);

    // xh = silu(causal_depthwise_conv(xz[:, :DI]))
    conv_silu_k<<<(unsigned)(((size_t)ROWS * DI) / 256), 256, 0, stream>>>(
        xzb, conv_w + i * DI * Kc, conv_b + i * DI, xhb);

    // x_dbl = xh @ x_proj_w^T  (N=96, edge-guarded)
    gemm_bf16<0><<<dim3((XD + 63) / 64, ROWS / 64), blk, 0, stream>>>(
        xhb, DI, x_proj_w + (size_t)i * XD * DI, DI, nullptr, nullptr,
        xdbl, XD, ROWS, XD, DI);

    // delta = softplus(x_dbl[:, :64] @ dt_proj_w^T + b); overwrite xz's
    // (dead) xh-half in place with ldc = 2*DI.
    gemm_bf16<2><<<dim3(DI / 64, ROWS / 64), blk, 0, stream>>>(
        xdbl, XD, dt_proj_w + (size_t)i * DI * DTR, DTR,
        dt_proj_b + i * DI, nullptr, xzb, 2 * DI, ROWS, DI, DTR);

    // selective scan + D skip + silu(z) gate -> xhb
    scan_k<<<Bb * (DI / 256), 256, 0, stream>>>(
        xzb, xhb, xdbl, A_log + i * DI * Ss, D_param + i * DI);

    // h = h + y @ out_proj_w^T   (residual fused, in-place on h)
    gemm_bf16<3><<<dim3(Dd / 64, ROWS / 64), blk, 0, stream>>>(
        xhb, DI, out_proj_w + (size_t)i * Dd * DI, DI, nullptr, h,
        h, Dd, ROWS, Dd, DI);
  }

  layernorm_k<<<ROWS, 256, 0, stream>>>(
      h, norm_f_w, norm_f_b, (float*)d_out, Dd);
}
